// SCSSystem_62646392980192
// MI455X (gfx1250) — matrix operand order repaired and checked by reading, not compiled, not run on hardware
//
#include <hip/hip_runtime.h>
#include <hip/hip_bf16.h>
#include <cstdint>

// ---------------- model constants ----------------
constexpr int   BATCH = 64;
constexpr int   NN    = 4096;            // nodes per region (64x64)
constexpr int   TT    = 32;              // CLK steps
constexpr int   BN    = BATCH * NN;      // 262144
constexpr int   KSPLIT = 4;              // split-K partials (deterministic)
constexpr float DECAY = 0.9f;
constexpr float THRESH = 1.0f;

typedef __attribute__((ext_vector_type(16))) _Float16 v16h;
typedef __attribute__((ext_vector_type(8)))  _Float16 v8h;
typedef __attribute__((ext_vector_type(4)))  _Float16 v4h;
typedef __attribute__((ext_vector_type(8)))  float    v8f;
typedef unsigned int __attribute__((ext_vector_type(4))) u32x4;
typedef int          __attribute__((ext_vector_type(8))) i32x8;
typedef int          __attribute__((ext_vector_type(4))) i32x4;

#if __has_builtin(__builtin_amdgcn_tensor_load_to_lds) && \
    __has_builtin(__builtin_amdgcn_s_wait_tensorcnt)
#define USE_TDM 1
#endif

// ---------------- f32 -> f16 weight conversion ----------------
__global__ void cvt_kernel(const float* __restrict__ src,
                           _Float16* __restrict__ dst, int n) {
  int i = 4 * (blockIdx.x * blockDim.x + threadIdx.x);
  if (i >= n) return;
  float4 f = *(const float4*)&src[i];
  v4h h; h[0] = (_Float16)f.x; h[1] = (_Float16)f.y;
         h[2] = (_Float16)f.z; h[3] = (_Float16)f.w;
  *(v4h*)&dst[i] = h;
}

// Build A_cat[4096][8192]: row i = [A_acc_to_pfc[i,:] | A_v1_to_pfc[i,:]]
__global__ void cvt_cat_kernel(const float* __restrict__ a2p,
                               const float* __restrict__ v2p,
                               _Float16* __restrict__ dst) {
  int i = 4 * (blockIdx.x * blockDim.x + threadIdx.x);
  if (i >= NN * NN) return;
  int row = i >> 12, col = i & 4095;
  float4 fa = *(const float4*)&a2p[i];
  float4 fv = *(const float4*)&v2p[i];
  v4h ha; ha[0] = (_Float16)fa.x; ha[1] = (_Float16)fa.y;
          ha[2] = (_Float16)fa.z; ha[3] = (_Float16)fa.w;
  v4h hv; hv[0] = (_Float16)fv.x; hv[1] = (_Float16)fv.y;
          hv[2] = (_Float16)fv.z; hv[3] = (_Float16)fv.w;
  size_t base = (size_t)row * 8192 + col;
  *(v4h*)&dst[base]        = ha;
  *(v4h*)&dst[base + 4096] = hv;
}

__global__ void init_kernel(float* __restrict__ v) {
  int i = blockIdx.x * blockDim.x + threadIdx.x;
  if (i < 3 * BN) v[i] = 0.0f;
}

// ---------------- phase 1: spikes (binary forward value) ----------------
__global__ void spike_kernel(const float* __restrict__ v,
                             _Float16* __restrict__ s_pfc,
                             _Float16* __restrict__ s_cat) {
  int i = blockIdx.x * blockDim.x + threadIdx.x;
  if (i >= BN) return;
  int b = i >> 12, n = i & 4095;
  float vp = v[i], va = v[BN + i], vv = v[2 * BN + i];
  s_pfc[i] = (_Float16)(vp > THRESH ? 1.0f : 0.0f);
  size_t j = (size_t)b * 8192 + n;
  s_cat[j]        = (_Float16)(va > THRESH ? 1.0f : 0.0f);
  s_cat[j + 4096] = (_Float16)(vv > THRESH ? 1.0f : 0.0f);
}

// ---------------- WMMA GEMM: C[kz][64][Nout] = S[64,K] @ A[Nout,K]^T -------
// Wave owns a 64(M)x16(N) strip -> each adjacency B-tile load feeds 4 WMMAs
// (64 FLOP per L2 byte; the 128MB of f16 weights stay L2-resident).
// S tile [64][KCHUNK] is DMAed into LDS by the Tensor Data Mover with
// hardware padding (pad_interval = 32 dwords, pad_amount = 4 dwords) to give
// the conflict-free 72-half row stride; double-buffered so the next tile's
// DMA overlaps the current tile's WMMA + B-load stream.
constexpr int KCHUNK  = 64;
constexpr int SSTRIDE = KCHUNK + 8;   // 72 halves = 144B rows

#ifdef USE_TDM
// Issue a TDM load of a [64 x KCHUNK] f16 tile (row stride K elements) from
// gsrc into LDS at lds_byte_addr, with per-row 16B padding.
__device__ __forceinline__ void tdm_load_tile(const _Float16* gsrc,
                                              unsigned lds_byte_addr, int K) {
  unsigned long long ga = (unsigned long long)(uintptr_t)gsrc;
  u32x4 g0;
  g0[0] = 1u;                                        // count=1 (valid), user
  g0[1] = lds_byte_addr;                             // lds_addr
  g0[2] = (unsigned)ga;                              // global_addr[31:0]
  g0[3] = (unsigned)((ga >> 32) & 0x01FFFFFFull)     // global_addr[56:32]
          | (2u << 30);                              // type = 2 ("image")
  unsigned long long q0 = (1ull << 16)               // data_size = 2 bytes
                        | (1ull << 20)               // pad_enable
                        | (4ull << 22)               // pad_interval: 32 dwords
                        | (3ull << 25)               // pad_amount: 4 dwords
                        | ((unsigned long long)(unsigned)K << 48); // t_dim0 lo
  unsigned long long q1 = ((unsigned long long)(unsigned)K >> 16)  // t_dim0 hi
                        | (64ull << 16)              // tensor_dim1 = 64 rows
                        | ((unsigned long long)KCHUNK << 48);      // tile_dim0
  unsigned long long q2 = 64ull                      // tile_dim1 = 64
                        | ((unsigned long long)(unsigned)K << 32); // d0_stride
  unsigned long long q3 = 0ull;                      // d1_stride = 0 (2D)
  i32x8 g1;
  g1[0] = (int)q0; g1[1] = (int)(q0 >> 32);
  g1[2] = (int)q1; g1[3] = (int)(q1 >> 32);
  g1[4] = (int)q2; g1[5] = (int)(q2 >> 32);
  g1[6] = (int)q3; g1[7] = (int)(q3 >> 32);
  i32x4 z4 = {0, 0, 0, 0};
  i32x8 z8 = {0, 0, 0, 0, 0, 0, 0, 0};
  __builtin_amdgcn_tensor_load_to_lds(g0, g1, z4, z4, z8, 0);
}
#endif

__global__ __launch_bounds__(256) void gemm_kernel(
    const _Float16* __restrict__ S,    // [64, K]
    const _Float16* __restrict__ A,    // [Nout, K] row-major
    float* __restrict__ Cp,            // [gridDim.y][64, Nout]
    int Nout, int K) {
  __shared__ _Float16 sS[2][64 * SSTRIDE];
  const int lane = threadIdx.x & 31;
  const int wave = threadIdx.x >> 5;
  const int n0   = blockIdx.x * 128 + wave * 16;
  const int kslen = K / (int)gridDim.y;
  const int kbeg  = blockIdx.y * kslen;
  const int nchunks = kslen / KCHUNK;

  v8f acc[4] = {};

  const int col = lane & 15;
  const int kbA = (lane >> 4) * 8;    // A-operand: K = kbA..+7, kbA+16..+23
  const int kbB = (lane >> 4) * 8;    // B-operand: K = kbB..+7, kbB+16..+23
  const _Float16* __restrict__ Brow = A + (size_t)(n0 + col) * K;

#ifdef USE_TDM
  // ---- TDM double-buffered pipeline ----
  if (wave == 0) {
    tdm_load_tile(S + kbeg, (unsigned)(uintptr_t)&sS[0][0], K);
    __builtin_amdgcn_s_wait_tensorcnt(0);
  }
  __syncthreads();
  for (int ci = 0; ci < nchunks; ++ci) {
    const int kc  = kbeg + ci * KCHUNK;
    const int cur = ci & 1;
    if (wave == 0 && ci + 1 < nchunks)
      tdm_load_tile(S + kc + KCHUNK, (unsigned)(uintptr_t)&sS[cur ^ 1][0], K);
    const _Float16* sBuf = &sS[cur][0];
#else
  // ---- fallback: register-staged pipelined fill ----
  const int c0 = threadIdx.x,       r0 = c0 >> 3, k80 = (c0 & 7) * 8;
  const int c1 = threadIdx.x + 256, r1 = c1 >> 3, k81 = (c1 & 7) * 8;
  v8h t0 = *(const v8h*)&S[(size_t)r0 * K + kbeg + k80];
  v8h t1 = *(const v8h*)&S[(size_t)r1 * K + kbeg + k81];
  for (int ci = 0; ci < nchunks; ++ci) {
    const int kc = kbeg + ci * KCHUNK;
    __syncthreads();
    *(v8h*)&sS[0][r0 * SSTRIDE + k80] = t0;
    *(v8h*)&sS[0][r1 * SSTRIDE + k81] = t1;
    __syncthreads();
    if (ci + 1 < nchunks) {
      t0 = *(const v8h*)&S[(size_t)r0 * K + kc + KCHUNK + k80];
      t1 = *(const v8h*)&S[(size_t)r1 * K + kc + KCHUNK + k81];
    }
    const _Float16* sBuf = &sS[0][0];
#endif
    #pragma unroll
    for (int kk = 0; kk < KCHUNK; kk += 32) {
      // B operand (adjacency^T tile) straight from L2: 32B per lane
      union { v16h v; v8h h[2]; } bu;
      bu.h[0] = *(const v8h*)&Brow[kc + kk + kbB];
      bu.h[1] = *(const v8h*)&Brow[kc + kk + kbB + 16];
      v16h bv = bu.v;
      #pragma unroll
      for (int m = 0; m < 4; ++m) {
        union { v16h v; v8h h[2]; } a;
        int r = m * 16 + col;
        a.h[0] = *(const v8h*)&sBuf[r * SSTRIDE + kk + kbA];
        a.h[1] = *(const v8h*)&sBuf[r * SSTRIDE + kk + kbA + 16];
        acc[m] = __builtin_amdgcn_wmma_f32_16x16x32_f16(
            false, a.v, false, bv, (short)0, acc[m], false, false);
      }
    }
#ifdef USE_TDM
    if (wave == 0) __builtin_amdgcn_s_wait_tensorcnt(0);
    __syncthreads();
  }
#else
  }
#endif

  // epilogue: C/D layout -> VGPR q: lanes 0-15 M=q, lanes 16-31 M=q+8
  float* __restrict__ Cz = Cp + (size_t)blockIdx.y * 64 * Nout;
  #pragma unroll
  for (int m = 0; m < 4; ++m) {
    #pragma unroll
    for (int q = 0; q < 8; ++q) {
      int row = m * 16 + q + (lane >> 4) * 8;
      Cz[(size_t)row * Nout + n0 + col] = acc[m][q];
    }
  }
}

// ---------------- phase 2: leaky integrate + reset-on-spike ----------------
// ax_av:  [KSPLIT][64][8192]  (cols 0..4095 -> ACC, 4096.. -> V1)
// ax_pfc: [KSPLIT][64][4096]
__global__ void update_kernel(float* __restrict__ v,
                              const float* __restrict__ ax_av,
                              const float* __restrict__ ax_pfc,
                              const float* __restrict__ ext_t) {
  int i = blockIdx.x * blockDim.x + threadIdx.x;
  if (i >= BN) return;
  int b = i >> 12, n = i & 4095;
  float vp = v[i], va = v[BN + i], vv = v[2 * BN + i];
  float sp = vp > THRESH ? 1.0f : 0.0f;
  float sa = va > THRESH ? 1.0f : 0.0f;
  float sv = vv > THRESH ? 1.0f : 0.0f;
  size_t j = (size_t)b * 8192 + n;
  float axp = 0.0f, axa = 0.0f, axv = 0.0f;
  #pragma unroll
  for (int p = 0; p < KSPLIT; ++p) {
    axp += ax_pfc[(size_t)p * BN + i];
    axa += ax_av[(size_t)p * 2 * BN + j];
    axv += ax_av[(size_t)p * 2 * BN + j + 4096];
  }
  v[i]          = DECAY * vp * (1.0f - sp) + axp + ext_t[i];
  v[BN + i]     = DECAY * va * (1.0f - sa) + axa;
  v[2 * BN + i] = DECAY * vv * (1.0f - sv) + axv;
}

// ---------------- host orchestration ----------------
// Workspace (~147 MB): f16 weights (128 MB, L2-resident in the hot loop),
// spikes, membranes, split-K partial buffers.
extern "C" void kernel_launch(void* const* d_in, const int* in_sizes, int n_in,
                              void* d_out, int out_size, void* d_ws, size_t ws_size,
                              hipStream_t stream) {
  const float* ext   = (const float*)d_in[0];  // [32,64,64,64]
  const float* A_p2a = (const float*)d_in[1];
  const float* A_a2p = (const float*)d_in[2];
  const float* A_p2v = (const float*)d_in[3];
  const float* A_v2p = (const float*)d_in[4];

  char* ws = (char*)d_ws;
  const size_t ADJ = (size_t)NN * NN;                 // 16,777,216 elements
  _Float16* A_stack = (_Float16*)(ws);                              // [8192][4096]
  _Float16* A_cat   = (_Float16*)(ws + 2 * ADJ * sizeof(_Float16)); // [4096][8192]
  size_t off = 4 * ADJ * sizeof(_Float16);
  _Float16* s_pfc   = (_Float16*)(ws + off); off += (size_t)BN * sizeof(_Float16);
  _Float16* s_cat   = (_Float16*)(ws + off); off += (size_t)2 * BN * sizeof(_Float16);
  float*    vbuf    = (float*)(ws + off);    off += (size_t)3 * BN * sizeof(float);
  float*    ax_av   = (float*)(ws + off);    off += (size_t)KSPLIT * 2 * BN * sizeof(float);
  float*    ax_pfc  = (float*)(ws + off);

  const int CVT_BLOCKS = (int)(ADJ / 4 / 256);        // 16384
  cvt_kernel    <<<CVT_BLOCKS, 256, 0, stream>>>(A_p2a, A_stack,        (int)ADJ);
  cvt_kernel    <<<CVT_BLOCKS, 256, 0, stream>>>(A_p2v, A_stack + ADJ,  (int)ADJ);
  cvt_cat_kernel<<<CVT_BLOCKS, 256, 0, stream>>>(A_a2p, A_v2p, A_cat);
  init_kernel   <<<(3 * BN) / 256, 256, 0, stream>>>(vbuf);

  for (int t = 0; t < TT; ++t) {
    spike_kernel<<<BN / 256, 256, 0, stream>>>(vbuf, s_pfc, s_cat);
    // [ax_acc | ax_v1] = s_pfc @ A_stack^T   (Nout=8192, K=4096)
    gemm_kernel<<<dim3(8192 / 128, KSPLIT), 256, 0, stream>>>(
        s_pfc, A_stack, ax_av, 8192, 4096);
    // ax_pfc = [s_acc|s_v1] @ A_cat^T        (Nout=4096, K=8192)
    gemm_kernel<<<dim3(4096 / 128, KSPLIT), 256, 0, stream>>>(
        s_cat, A_cat, ax_pfc, 4096, 8192);
    update_kernel<<<BN / 256, 256, 0, stream>>>(
        vbuf, ax_av, ax_pfc, ext + (size_t)t * BN);
  }

  // output: final v_pfc [64,64,64] f32
  (void)hipMemcpyAsync(d_out, vbuf, (size_t)BN * sizeof(float),
                       hipMemcpyDeviceToDevice, stream);
}
